// CrossEncoderLongformer_45320494907768
// MI455X (gfx1250) — matrix operand order repaired and checked by reading, not compiled, not run on hardware
//
#include <hip/hip_runtime.h>
#include <hip/hip_bf16.h>
#include <math.h>

typedef _Float16 half_t;
typedef __attribute__((ext_vector_type(16))) _Float16 v16h;
typedef __attribute__((ext_vector_type(8)))  _Float16 v8h;
typedef __attribute__((ext_vector_type(8)))  float    v8f;

constexpr int kB   = 2;
constexpr int kL   = 4096;
constexpr int kD   = 768;
constexpr int kNH  = 12;
constexpr int kHD  = 64;
constexpr int kNL  = 2;
constexpr int kDFF = 3072;
constexpr int kG   = 80;
constexpr int kC   = 16;
constexpr int kDH  = 384;
constexpr int kWIN = 256;
constexpr int kM   = kB * kL;

// ---------------------------------------------------------------------------
// WMMA fragment helpers. Layouts chosen so every fragment is 2x ds_load_b128.
// ---------------------------------------------------------------------------
__device__ inline v8h v8h_zero() {
  v8h z;
#pragma unroll
  for (int j = 0; j < 8; ++j) z[j] = (half_t)0.f;
  return z;
}
__device__ inline v16h concat8(v8h lo, v8h hi) {
  return __builtin_shufflevector(lo, hi, 0, 1, 2, 3, 4, 5, 6, 7, 8, 9, 10, 11,
                                 12, 13, 14, 15);
}
// A matrix 16x32 (MxK), K-contiguous rows, lda in halves (multiple of 8).
__device__ inline v16h frag_a_fast(const half_t* src, int lda) {
  int lane = threadIdx.x & 31;
  int m = lane & 15, g = lane >> 4;
  const half_t* p = src + m * lda + 8 * g;
  v8h lo = *(const v8h*)(p);
  v8h hi = *(const v8h*)(p + 16);
  return concat8(lo, hi);
}
// B matrix 32x16 (KxN) staged transposed: element (k,n) = src[n*ldn + k].
__device__ inline v16h frag_bt_fast(const half_t* src, int ldn) {
  int lane = threadIdx.x & 31;
  int n = lane & 15, g = lane >> 4;
  const half_t* p = src + n * ldn + 8 * g;
  v8h lo = *(const v8h*)(p);
  v8h hi = *(const v8h*)(p + 16);
  return concat8(lo, hi);
}
__device__ inline v8f wmma16(v16h a, v16h b, v8f c) {
  return __builtin_amdgcn_wmma_f32_16x16x32_f16(false, a, false, b, (short)0, c,
                                                false, false);
}
// CDNA5 async global->LDS 16B copy (ASYNCcnt-tracked); lds_off = low 32 bits
// of the generic LDS address.
__device__ inline void async_load_lds_b128(unsigned lds_off,
                                           const half_t* gaddr) {
  asm volatile("global_load_async_to_lds_b128 %0, %1, off"
               :
               : "v"(lds_off), "v"(gaddr)
               : "memory");
}
__device__ inline void wait_asynccnt0() {
  asm volatile("s_wait_asynccnt 0" ::: "memory");
}

// ---------------------------------------------------------------------------
// Elementwise / conversion kernels (128-bit vectorized)
// ---------------------------------------------------------------------------
__global__ void f32_to_f16_kernel(const float* __restrict__ in,
                                  half_t* __restrict__ out, size_t n) {
  size_t i8 = ((size_t)blockIdx.x * blockDim.x + threadIdx.x) * 8;
  size_t stride = (size_t)gridDim.x * blockDim.x * 8;
  for (; i8 + 8 <= n; i8 += stride) {
    float4 a = *(const float4*)(in + i8);
    float4 b = *(const float4*)(in + i8 + 4);
    v8h o;
    o[0] = (half_t)a.x; o[1] = (half_t)a.y; o[2] = (half_t)a.z; o[3] = (half_t)a.w;
    o[4] = (half_t)b.x; o[5] = (half_t)b.y; o[6] = (half_t)b.z; o[7] = (half_t)b.w;
    *(v8h*)(out + i8) = o;
  }
  if (blockIdx.x == 0 && threadIdx.x == 0) {
    for (size_t t = n & ~(size_t)7; t < n; ++t) out[t] = (half_t)in[t];
  }
}

__global__ void gather_rows_f16_kernel(const float* __restrict__ src,
                                       const int* __restrict__ idx,
                                       half_t* __restrict__ dst, int nrows,
                                       int nb, int srcL) {
  long i = (long)blockIdx.x * 256 + threadIdx.x;  // v8h index
  long total = (long)nb * nrows * (kD / 8);
  if (i >= total) return;
  int cb = (int)(i % (kD / 8)) * 8;
  long rr = i / (kD / 8);
  int r = (int)(rr % nrows);
  int b = (int)(rr / nrows);
  const float* s = &src[((long)b * srcL + idx[r]) * kD + cb];
  float4 a = *(const float4*)(s);
  float4 bb = *(const float4*)(s + 4);
  v8h o;
  o[0] = (half_t)a.x;  o[1] = (half_t)a.y;  o[2] = (half_t)a.z;  o[3] = (half_t)a.w;
  o[4] = (half_t)bb.x; o[5] = (half_t)bb.y; o[6] = (half_t)bb.z; o[7] = (half_t)bb.w;
  *(v8h*)(&dst[i * 8]) = o;
}

// ---------------------------------------------------------------------------
// Embedding + LayerNorm  and  residual-add + LayerNorm
// ---------------------------------------------------------------------------
__global__ __launch_bounds__(256) void embed_ln_kernel(
    const int* __restrict__ ids, const float* __restrict__ ew,
    const float* __restrict__ ep, const float* __restrict__ gam,
    const float* __restrict__ bet, float* __restrict__ hf,
    half_t* __restrict__ hh) {
  long row = blockIdx.x;
  int l = (int)(row % kL);
  int tid = threadIdx.x;
  int id = ids[row];
  __shared__ float xs[kD];
  __shared__ float red[256];
  float local = 0.f;
  for (int c = tid; c < kD; c += 256) {
    float x = ew[(long)id * kD + c] + ep[(long)l * kD + c];
    xs[c] = x;
    local += x;
  }
  red[tid] = local;
  __syncthreads();
  for (int s = 128; s > 0; s >>= 1) {
    if (tid < s) red[tid] += red[tid + s];
    __syncthreads();
  }
  float mu = red[0] / kD;
  __syncthreads();
  local = 0.f;
  for (int c = tid; c < kD; c += 256) {
    float dx = xs[c] - mu;
    local += dx * dx;
  }
  red[tid] = local;
  __syncthreads();
  for (int s = 128; s > 0; s >>= 1) {
    if (tid < s) red[tid] += red[tid + s];
    __syncthreads();
  }
  float rstd = rsqrtf(red[0] / kD + 1e-5f);
  for (int c = tid; c < kD; c += 256) {
    float y = (xs[c] - mu) * rstd * gam[c] + bet[c];
    hf[row * kD + c] = y;
    hh[row * kD + c] = (half_t)y;
  }
}

__global__ __launch_bounds__(256) void add_ln_kernel(
    float* __restrict__ hf, const float* __restrict__ add,
    const float* __restrict__ gam, const float* __restrict__ bet,
    half_t* __restrict__ hh) {
  long row = blockIdx.x;
  int tid = threadIdx.x;
  __shared__ float xs[kD];
  __shared__ float red[256];
  float local = 0.f;
  for (int c = tid; c < kD; c += 256) {
    float x = hf[row * kD + c] + add[row * kD + c];
    xs[c] = x;
    local += x;
  }
  red[tid] = local;
  __syncthreads();
  for (int s = 128; s > 0; s >>= 1) {
    if (tid < s) red[tid] += red[tid + s];
    __syncthreads();
  }
  float mu = red[0] / kD;
  __syncthreads();
  local = 0.f;
  for (int c = tid; c < kD; c += 256) {
    float dx = xs[c] - mu;
    local += dx * dx;
  }
  red[tid] = local;
  __syncthreads();
  for (int s = 128; s > 0; s >>= 1) {
    if (tid < s) red[tid] += red[tid + s];
    __syncthreads();
  }
  float rstd = rsqrtf(red[0] / kD + 1e-5f);
  for (int c = tid; c < kD; c += 256) {
    float y = (xs[c] - mu) * rstd * gam[c] + bet[c];
    hf[row * kD + c] = y;
    hh[row * kD + c] = (half_t)y;
  }
}

// ---------------------------------------------------------------------------
// Tiled WMMA GEMM with async-LDS double buffering.
// Block tile 256x64, K-step 32, 8 waves; wave owns 32x64 (8 WMMA / K-step).
// A tile staged via GLOBAL_LOAD_ASYNC_TO_LDS_B128 (ASYNCcnt), ping-pong LDS;
// B tile staged transposed through VGPRs (fragments = 2x ds_load_b128).
// GUARD=false assumes M % 256 == 0 (no row predication anywhere).
// ---------------------------------------------------------------------------
template <bool GUARD>
__global__ __launch_bounds__(256) void gemm_wmma_kernel(
    const half_t* __restrict__ A, const half_t* __restrict__ Bm,
    const float* __restrict__ bias, float* __restrict__ Cf,
    half_t* __restrict__ Ch, int M, int N, int K, int act) {
  __shared__ __align__(16) half_t Als[2][256][40];
  __shared__ __align__(16) half_t Bt[2][64][40];  // transposed: [n][k]
  int m0 = blockIdx.y * 256;
  int n0 = blockIdx.x * 64;
  int tid = threadIdx.x;
  int wave = tid >> 5, lane = tid & 31;
  v8f acc[2][4];
  for (int rt = 0; rt < 2; ++rt)
    for (int nt = 0; nt < 4; ++nt)
      for (int j = 0; j < 8; ++j) acc[rt][nt][j] = 0.f;

  const int nk = K / 32;

  auto stage = [&](int kt, int buf) {
    int k0 = kt * 32;
    // A tile: 256x32 halves = 1024 x 16B, 4 async copies per thread
#pragma unroll
    for (int i = 0; i < 4; ++i) {
      int idx = tid + 256 * i;
      int m = idx >> 2, kc = (idx & 3) * 8;
      half_t* lp = &Als[buf][m][kc];
      unsigned lo = (unsigned)(unsigned long long)lp;
      if (!GUARD || (m0 + m < M)) {
        async_load_lds_b128(lo, A + (long)(m0 + m) * K + k0 + kc);
      } else {
        *(v8h*)lp = v8h_zero();
      }
    }
    // B tile: 32x64 = 256 v8h, one per thread, transposed scatter to LDS
    int kk = tid >> 3, nc = (tid & 7) * 8;
    v8h v = *(const v8h*)(Bm + (long)(k0 + kk) * N + n0 + nc);
#pragma unroll
    for (int j = 0; j < 8; ++j) Bt[buf][nc + j][kk] = v[j];
    if (kt + 1 < nk)
      __builtin_prefetch(Bm + (long)(kt * 32 + 32 + kk) * N + n0, 0, 1);
  };

  stage(0, 0);
  for (int kt = 0; kt < nk; ++kt) {
    int buf = kt & 1;
    wait_asynccnt0();   // this wave's async copies for tile kt done
    __syncthreads();    // all waves' copies + B ds_stores visible
    if (kt + 1 < nk) stage(kt + 1, buf ^ 1);
    v16h bf[4];
#pragma unroll
    for (int nt = 0; nt < 4; ++nt)
      bf[nt] = frag_bt_fast(&Bt[buf][nt * 16][0], 40);
#pragma unroll
    for (int rt = 0; rt < 2; ++rt) {
      v16h af = frag_a_fast(&Als[buf][wave * 32 + rt * 16][0], 40);
#pragma unroll
      for (int nt = 0; nt < 4; ++nt)
        acc[rt][nt] = wmma16(af, bf[nt], acc[rt][nt]);
    }
  }
  int g = lane >> 4, nl = lane & 15;
  for (int rt = 0; rt < 2; ++rt)
    for (int nt = 0; nt < 4; ++nt) {
      int col = n0 + nt * 16 + nl;
      float bb = bias ? bias[col] : 0.f;
#pragma unroll
      for (int i = 0; i < 8; ++i) {
        int row = m0 + wave * 32 + rt * 16 + i + 8 * g;
        if (!GUARD || row < M) {
          float v = acc[rt][nt][i] + bb;
          if (act == 1) v = 0.5f * v * (1.f + erff(v * 0.70710678118f));
          if (Cf) Cf[(long)row * N + col] = v;
          if (Ch) Ch[(long)row * N + col] = (half_t)v;
        }
      }
    }
}

// ---------------------------------------------------------------------------
// Band attention (flash-style, WMMA): block = (chunk n, head h, batch b).
// ---------------------------------------------------------------------------
struct __align__(16) BandSmem {
  union {
    half_t q[256][72];  // Q chunk (dead after fragment preload)
    struct {
      half_t K[32][72];
      half_t V[64][40];     // transposed: [d][key]
      half_t P[8][32][40];  // per-wave probability staging
    } s;
  };
};

__global__ __launch_bounds__(256) void band_attn_kernel(
    const half_t* __restrict__ Q, const half_t* __restrict__ Kb,
    const half_t* __restrict__ Vb, const int* __restrict__ gidx,
    float* __restrict__ Out) {
  __shared__ BandSmem sm;
  int n = blockIdx.x, h = blockIdx.y, b = blockIdx.z;
  int tid = threadIdx.x, wave = tid >> 5, lane = tid & 31;
  int gq = lane >> 4, nl = lane & 15;
  long base = ((long)b * kL + (long)n * 256) * kD + h * kHD;

#pragma unroll
  for (int i = 0; i < 8; ++i) {
    int idx = tid + 256 * i;
    int r = idx >> 3, c = (idx & 7) * 8;
    v8h v = *(const v8h*)(Q + base + (long)r * kD + c);
#pragma unroll
    for (int j = 0; j < 8; ++j) v[j] = v[j] * (half_t)0.125f;
    *(v8h*)(&sm.q[r][c]) = v;
  }
  __syncthreads();
  v16h qf[2][2];
  for (int rt = 0; rt < 2; ++rt)
    for (int ks = 0; ks < 2; ++ks)
      qf[rt][ks] = frag_a_fast(&sm.q[wave * 32 + rt * 16][ks * 32], 72);

  float mrow[2][8], lrow[2][8];
  v8f oacc[2][4];
  for (int rt = 0; rt < 2; ++rt) {
    for (int i = 0; i < 8; ++i) {
      mrow[rt][i] = -3e38f;
      lrow[rt][i] = 0.f;
    }
    for (int ct = 0; ct < 4; ++ct)
      for (int i = 0; i < 8; ++i) oacc[rt][ct][i] = 0.f;
  }

  for (int t = 0; t < 27; ++t) {
    __syncthreads();
    {
      int kr = tid >> 3, c = (tid & 7) * 8;
      v8h kv = v8h_zero(), vv = v8h_zero();
      long grow = -1;
      if (t < 3) {
        int gi = t * 32 + kr;
        if (gi < kG) grow = (long)b * kL + gidx[gi];
      } else {
        int m = (t - 3) * 32 + kr;
        int j = n * 256 + m - 256;
        if (j >= 0 && j < kL) grow = (long)b * kL + j;
      }
      if (grow >= 0) {
        long off = grow * kD + h * kHD + c;
        kv = *(const v8h*)(Kb + off);
        vv = *(const v8h*)(Vb + off);
      }
      *(v8h*)(&sm.s.K[kr][c]) = kv;
#pragma unroll
      for (int j = 0; j < 8; ++j) sm.s.V[c + j][kr] = vv[j];  // transpose
    }
    __syncthreads();

    for (int rt = 0; rt < 2; ++rt) {
      v8f s[2];
      for (int n2 = 0; n2 < 2; ++n2)
        for (int i = 0; i < 8; ++i) s[n2][i] = 0.f;
      for (int n2 = 0; n2 < 2; ++n2)
        for (int ks = 0; ks < 2; ++ks)
          s[n2] = wmma16(qf[rt][ks],
                         frag_bt_fast(&sm.s.K[n2 * 16][ks * 32], 72), s[n2]);
      for (int n2 = 0; n2 < 2; ++n2) {
        int kl = n2 * 16 + nl;
#pragma unroll
        for (int i = 0; i < 8; ++i) {
          int ci = wave * 32 + rt * 16 + i + 8 * gq;
          bool valid;
          if (t < 3) {
            valid = (t * 32 + kl) < kG;
          } else {
            int m = (t - 3) * 32 + kl;
            int j = n * 256 + m - 256;
            int off = m - 256 - ci;
            valid = (off >= -kWIN) && (off <= kWIN) && (j >= 0) && (j < kL);
          }
          if (!valid) s[n2][i] = -1.0e9f;
        }
      }
      float mnew[8], alpha[8], rsum[8];
#pragma unroll
      for (int i = 0; i < 8; ++i) {
        float v = fmaxf(s[0][i], s[1][i]);
        for (int msk = 1; msk < 16; msk <<= 1)
          v = fmaxf(v, __shfl_xor(v, msk, 16));
        mnew[i] = fmaxf(mrow[rt][i], v);
        alpha[i] = expf(mrow[rt][i] - mnew[i]);
        rsum[i] = 0.f;
      }
      for (int n2 = 0; n2 < 2; ++n2)
#pragma unroll
        for (int i = 0; i < 8; ++i) {
          float pv = expf(s[n2][i] - mnew[i]);
          s[n2][i] = pv;
          rsum[i] += pv;
        }
#pragma unroll
      for (int i = 0; i < 8; ++i) {
        float rs = rsum[i];
        for (int msk = 1; msk < 16; msk <<= 1) rs += __shfl_xor(rs, msk, 16);
        lrow[rt][i] = lrow[rt][i] * alpha[i] + rs;
        mrow[rt][i] = mnew[i];
      }
      for (int ct = 0; ct < 4; ++ct)
#pragma unroll
        for (int i = 0; i < 8; ++i) oacc[rt][ct][i] *= alpha[i];
      for (int n2 = 0; n2 < 2; ++n2)
#pragma unroll
        for (int i = 0; i < 8; ++i)
          sm.s.P[wave][rt * 16 + i + 8 * gq][n2 * 16 + nl] = (half_t)s[n2][i];
      v16h pf = frag_a_fast(&sm.s.P[wave][rt * 16][0], 40);
      for (int ct = 0; ct < 4; ++ct)
        oacc[rt][ct] =
            wmma16(pf, frag_bt_fast(&sm.s.V[ct * 16][0], 40), oacc[rt][ct]);
    }
  }
  for (int rt = 0; rt < 2; ++rt)
    for (int ct = 0; ct < 4; ++ct)
#pragma unroll
      for (int i = 0; i < 8; ++i) {
        int r = wave * 32 + rt * 16 + i + 8 * gq;
        Out[base + (long)r * kD + ct * 16 + nl] = oacc[rt][ct][i] / lrow[rt][i];
      }
}

// ---------------------------------------------------------------------------
// Global-token attention: block = (g, h, b); full-L softmax; overwrites the
// band output rows at g_idx positions.
// ---------------------------------------------------------------------------
__global__ __launch_bounds__(256) void global_attn_kernel(
    const half_t* __restrict__ QG, const half_t* __restrict__ KA,
    const half_t* __restrict__ VA, const int* __restrict__ gidx,
    float* __restrict__ Out) {
  int g = blockIdx.x, h = blockIdx.y, b = blockIdx.z;
  __shared__ float sc[kL];
  __shared__ float red[256];
  __shared__ float qv[kHD];
  __shared__ float outv[kHD];
  int tid = threadIdx.x;
  if (tid < kHD) {
    qv[tid] = (float)QG[((long)b * kG + g) * kD + h * kHD + tid] * 0.125f;
    outv[tid] = 0.f;
  }
  __syncthreads();
  float lmax = -3e38f;
  for (int j = tid; j < kL; j += 256) {
    const v8h* kp = (const v8h*)&KA[((long)b * kL + j) * kD + h * kHD];
    float s = 0.f;
#pragma unroll
    for (int cc = 0; cc < 8; ++cc) {
      v8h kv = kp[cc];
#pragma unroll
      for (int u = 0; u < 8; ++u) s += qv[cc * 8 + u] * (float)kv[u];
    }
    sc[j] = s;
    lmax = fmaxf(lmax, s);
  }
  red[tid] = lmax;
  __syncthreads();
  for (int st = 128; st > 0; st >>= 1) {
    if (tid < st) red[tid] = fmaxf(red[tid], red[tid + st]);
    __syncthreads();
  }
  float mx = red[0];
  __syncthreads();
  float lsum = 0.f;
  for (int j = tid; j < kL; j += 256) {
    float p = expf(sc[j] - mx);
    sc[j] = p;
    lsum += p;
  }
  red[tid] = lsum;
  __syncthreads();
  for (int st = 128; st > 0; st >>= 1) {
    if (tid < st) red[tid] += red[tid + st];
    __syncthreads();
  }
  float denom = red[0];
  __syncthreads();
  float oa[kHD];
#pragma unroll
  for (int c = 0; c < kHD; ++c) oa[c] = 0.f;
  for (int j = tid; j < kL; j += 256) {
    float p = sc[j];
    const v8h* vp = (const v8h*)&VA[((long)b * kL + j) * kD + h * kHD];
#pragma unroll
    for (int cc = 0; cc < 8; ++cc) {
      v8h vv = vp[cc];
#pragma unroll
      for (int u = 0; u < 8; ++u) oa[cc * 8 + u] += p * (float)vv[u];
    }
  }
#pragma unroll
  for (int c = 0; c < kHD; ++c) atomicAdd(&outv[c], oa[c]);  // ds_add_f32
  __syncthreads();
  int row = gidx[g];
  for (int c = tid; c < kHD; c += 256)
    Out[((long)b * kL + row) * kD + h * kHD + c] = outv[c] / denom;
}

// ---------------------------------------------------------------------------
// Final head: logits[r] = dot(gelu'd H1 row, hW2) + hb2
// ---------------------------------------------------------------------------
__global__ void head_out_kernel(const half_t* __restrict__ H1,
                                const float* __restrict__ W2,
                                const float* __restrict__ b2,
                                float* __restrict__ out) {
  int r = threadIdx.x;
  if (r >= kB * kC) return;
  float s = b2[0];
  for (int c = 0; c < kDH; ++c) s += (float)H1[(long)r * kDH + c] * W2[c];
  out[r] = s;
}

// ---------------------------------------------------------------------------
// Host orchestration
// ---------------------------------------------------------------------------
extern "C" void kernel_launch(void* const* d_in, const int* in_sizes, int n_in,
                              void* d_out, int out_size, void* d_ws,
                              size_t ws_size, hipStream_t stream) {
  (void)in_sizes; (void)n_in; (void)out_size; (void)ws_size;
  const int* ids  = (const int*)d_in[0];
  const int* cand = (const int*)d_in[1];
  const int* gidx = (const int*)d_in[2];
  const float* embw = (const float*)d_in[3];
  const float* embp = (const float*)d_in[4];
  const float* lneg = (const float*)d_in[5];
  const float* lneb = (const float*)d_in[6];
  const float* Wq = (const float*)d_in[7];   const float* bq = (const float*)d_in[8];
  const float* Wk = (const float*)d_in[9];   const float* bk = (const float*)d_in[10];
  const float* Wv = (const float*)d_in[11];  const float* bv = (const float*)d_in[12];
  const float* Wqg = (const float*)d_in[13]; const float* bqg = (const float*)d_in[14];
  const float* Wkg = (const float*)d_in[15]; const float* bkg = (const float*)d_in[16];
  const float* Wvg = (const float*)d_in[17]; const float* bvg = (const float*)d_in[18];
  const float* Wo = (const float*)d_in[19];  const float* bo = (const float*)d_in[20];
  const float* ln1g = (const float*)d_in[21]; const float* ln1b = (const float*)d_in[22];
  const float* Wf1 = (const float*)d_in[23]; const float* bf1 = (const float*)d_in[24];
  const float* Wf2 = (const float*)d_in[25]; const float* bf2 = (const float*)d_in[26];
  const float* ln2g = (const float*)d_in[27]; const float* ln2b = (const float*)d_in[28];
  const float* hW1 = (const float*)d_in[29]; const float* hb1 = (const float*)d_in[30];
  const float* hW2 = (const float*)d_in[31]; const float* hb2 = (const float*)d_in[32];

  char* wp = (char*)d_ws;
  auto alloc = [&](size_t bytes) -> void* {
    void* r = (void*)wp;
    wp += (bytes + 255) & ~(size_t)255;
    return r;
  };
  float*  hf  = (float*)alloc((size_t)kM * kD * 4);
  float*  ha  = (float*)alloc((size_t)kM * kD * 4);   // attn out (f32)
  float*  ho  = (float*)alloc((size_t)kM * kD * 4);   // proj/FFN out (f32)
  half_t* hh  = (half_t*)alloc((size_t)kM * kD * 2);
  half_t* qh  = (half_t*)alloc((size_t)kM * kD * 2);
  half_t* kh  = (half_t*)alloc((size_t)kM * kD * 2);
  half_t* vh  = (half_t*)alloc((size_t)kM * kD * 2);
  half_t* kgh = (half_t*)alloc((size_t)kM * kD * 2);
  half_t* vgh = (half_t*)alloc((size_t)kM * kD * 2);
  half_t* ffh = (half_t*)alloc((size_t)kM * kDFF * 2);
  half_t* w16 = (half_t*)alloc((size_t)kD * kDFF * 2);  // largest weight slab
  half_t* xgh = (half_t*)alloc((size_t)kB * kG * kD * 2);
  half_t* qgh = (half_t*)alloc((size_t)kB * kG * kD * 2);
  half_t* hch = (half_t*)alloc((size_t)kB * kC * kD * 2);
  half_t* h1h = (half_t*)alloc((size_t)kB * kC * kDH * 2);

  auto conv = [&](const float* s, half_t* d, size_t n) {
    int grid = (int)((n / 8 + 255) / 256);
    if (grid > 16384) grid = 16384;
    if (grid < 1) grid = 1;
    f32_to_f16_kernel<<<grid, 256, 0, stream>>>(s, d, n);
  };
  auto gemm = [&](const half_t* A, const half_t* Bw, const float* bias,
                  float* Cf, half_t* Ch, int M, int N, int K, int act) {
    dim3 g(N / 64, (M + 255) / 256);
    if (M % 256 == 0)
      gemm_wmma_kernel<false><<<g, 256, 0, stream>>>(A, Bw, bias, Cf, Ch, M, N,
                                                     K, act);
    else
      gemm_wmma_kernel<true><<<g, 256, 0, stream>>>(A, Bw, bias, Cf, Ch, M, N,
                                                    K, act);
  };

  embed_ln_kernel<<<kM, 256, 0, stream>>>(ids, embw, embp, lneg, lneb, hf, hh);

  const size_t DD = (size_t)kD * kD;
  for (int l = 0; l < kNL; ++l) {
    conv(Wq + l * DD, w16, DD);
    gemm(hh, w16, bq + l * kD, nullptr, qh, kM, kD, kD, 0);
    conv(Wk + l * DD, w16, DD);
    gemm(hh, w16, bk + l * kD, nullptr, kh, kM, kD, kD, 0);
    conv(Wv + l * DD, w16, DD);
    gemm(hh, w16, bv + l * kD, nullptr, vh, kM, kD, kD, 0);
    band_attn_kernel<<<dim3(kL / 256, kNH, kB), 256, 0, stream>>>(qh, kh, vh,
                                                                  gidx, ha);
    conv(Wkg + l * DD, w16, DD);
    gemm(hh, w16, bkg + l * kD, nullptr, kgh, kM, kD, kD, 0);
    conv(Wvg + l * DD, w16, DD);
    gemm(hh, w16, bvg + l * kD, nullptr, vgh, kM, kD, kD, 0);
    gather_rows_f16_kernel<<<((kB * kG * kD / 8) + 255) / 256, 256, 0, stream>>>(
        hf, gidx, xgh, kG, kB, kL);
    conv(Wqg + l * DD, w16, DD);
    gemm(xgh, w16, bqg + l * kD, nullptr, qgh, kB * kG, kD, kD, 0);
    global_attn_kernel<<<dim3(kG, kNH, kB), 256, 0, stream>>>(qgh, kgh, vgh,
                                                              gidx, ha);
    conv(ha, qh, (size_t)kM * kD);  // attn out -> f16 (qh reused)
    conv(Wo + l * DD, w16, DD);
    gemm(qh, w16, bo + l * kD, ho, nullptr, kM, kD, kD, 0);
    add_ln_kernel<<<kM, 256, 0, stream>>>(hf, ho, ln1g + l * kD, ln1b + l * kD,
                                          hh);
    conv(Wf1 + (size_t)l * kD * kDFF, w16, (size_t)kD * kDFF);
    gemm(hh, w16, bf1 + l * kDFF, nullptr, ffh, kM, kDFF, kD, 1);  // GELU
    conv(Wf2 + (size_t)l * kDFF * kD, w16, (size_t)kDFF * kD);
    gemm(ffh, w16, bf2 + l * kD, ho, nullptr, kM, kD, kDFF, 0);
    add_ln_kernel<<<kM, 256, 0, stream>>>(hf, ho, ln2g + l * kD, ln2b + l * kD,
                                          hh);
  }

  gather_rows_f16_kernel<<<((kB * kC * kD / 8) + 255) / 256, 256, 0, stream>>>(
      hf, cand, hch, kC, kB, kL);
  conv(hW1, w16, (size_t)kD * kDH);
  gemm(hch, w16, hb1, nullptr, h1h, kB * kC, kDH, kD, 1);  // GELU
  head_out_kernel<<<1, 64, 0, stream>>>(h1h, hW2, hb2, (float*)d_out);
}
